// VectorQuantizer_78451872629292
// MI455X (gfx1250) — matrix operand order repaired and checked by reading, not compiled, not run on hardware
//
#include <hip/hip_runtime.h>
#include <hip/hip_bf16.h>

typedef __attribute__((ext_vector_type(16))) __bf16 v16bf;
typedef __attribute__((ext_vector_type(8)))  __bf16 v8bf;
typedef __attribute__((ext_vector_type(8)))  float  v8f;
typedef __attribute__((ext_vector_type(4)))  float  v4f;   // native vector: OK for NT builtins

#define DIM         64
#define KCB         1024
#define ROWS_PER_WG 128
#define PHASE_ROWS  256
#define LDS_STRIDE  72   // 64 bf16 + 8 pad -> 144B row stride, conflict-friendly

__device__ __forceinline__ v16bf pack16(v4f p0, v4f p1, v4f p2, v4f p3) {
    v16bf v;
    v[0]=(__bf16)p0.x;  v[1]=(__bf16)p0.y;  v[2]=(__bf16)p0.z;  v[3]=(__bf16)p0.w;
    v[4]=(__bf16)p1.x;  v[5]=(__bf16)p1.y;  v[6]=(__bf16)p1.z;  v[7]=(__bf16)p1.w;
    v[8]=(__bf16)p2.x;  v[9]=(__bf16)p2.y;  v[10]=(__bf16)p2.z; v[11]=(__bf16)p2.w;
    v[12]=(__bf16)p3.x; v[13]=(__bf16)p3.y; v[14]=(__bf16)p3.z; v[15]=(__bf16)p3.w;
    return v;
}

__global__ __launch_bounds__(256)
void vq_wmma_kernel(const float* __restrict__ x, const float* __restrict__ cb,
                    float* __restrict__ disc, float* __restrict__ quant)
{
    __shared__ __bf16 ldsC[PHASE_ROWS * LDS_STRIDE]; // 36 KB staged bf16 codebook
    __shared__ float  cnormS[PHASE_ROWS];            // ||c||^2 per staged row
    __shared__ int    idxS[ROWS_PER_WG];             // per-row argmin

    const int t     = threadIdx.x;
    const int wave  = t >> 5;
    const int lane  = t & 31;
    const int lhalf = lane >> 4;   // 0/1 : which K-half / which M-half of C
    const int l15   = lane & 15;
    const int rowBase = blockIdx.x * ROWS_PER_WG;

    // ---- A-matrix: 16 rows of x per wave, CDNA5 16-bit A layout ----
    // lane l<16 holds row l (K in {0..7,16..23}), lane l>=16 holds row l-16
    // (K in {8..15,24..31}); a1 covers K 32..63 with the same pattern.
    const int myRow = rowBase + wave * 16 + l15;
    const v4f* xr = (const v4f*)(x + (size_t)myRow * DIM);
    // x is read exactly once per call: non-temporal loads keep it out of L2.
    v4f p0 = __builtin_nontemporal_load(&xr[2*lhalf]);
    v4f p1 = __builtin_nontemporal_load(&xr[2*lhalf+1]);
    v4f p2 = __builtin_nontemporal_load(&xr[4+2*lhalf]);
    v4f p3 = __builtin_nontemporal_load(&xr[5+2*lhalf]);
    v4f q0 = __builtin_nontemporal_load(&xr[8+2*lhalf]);
    v4f q1 = __builtin_nontemporal_load(&xr[9+2*lhalf]);
    v4f q2 = __builtin_nontemporal_load(&xr[12+2*lhalf]);
    v4f q3 = __builtin_nontemporal_load(&xr[13+2*lhalf]);
    v16bf a0 = pack16(p0, p1, p2, p3);
    v16bf a1 = pack16(q0, q1, q2, q3);

    float bestV[8];
    int   bestI[8];
#pragma unroll
    for (int r = 0; r < 8; ++r) { bestV[r] = 3.402823466e38f; bestI[r] = 0; }

    for (int p = 0; p < KCB / PHASE_ROWS; ++p) {
        __syncthreads();  // previous phase readers done before restaging
        // ---- stage 256 codebook rows as bf16 + squared norms ----
        {
            const v4f* c4 = (const v4f*)(cb + (size_t)(p * PHASE_ROWS + t) * DIM);
            __bf16* dst = &ldsC[t * LDS_STRIDE];
            float ss = 0.f;
#pragma unroll
            for (int j = 0; j < 16; j += 2) {
                v4f u = c4[j], w = c4[j + 1];
                ss += u.x*u.x + u.y*u.y + u.z*u.z + u.w*u.w
                    + w.x*w.x + w.y*w.y + w.z*w.z + w.w*w.w;
                v8bf b;
                b[0]=(__bf16)u.x; b[1]=(__bf16)u.y; b[2]=(__bf16)u.z; b[3]=(__bf16)u.w;
                b[4]=(__bf16)w.x; b[5]=(__bf16)w.y; b[6]=(__bf16)w.z; b[7]=(__bf16)w.w;
                *(v8bf*)(dst + j * 4) = b;
            }
            cnormS[t] = ss;
        }
        __syncthreads();

        // ---- 16 column-tiles per phase; D=64 reduction = 2 WMMA ----
        for (int cbt = 0; cbt < PHASE_ROWS / 16; ++cbt) {
            const int lr = cbt * 16 + l15;                    // my B column (staged idx)
            const __bf16* src = &ldsC[lr * LDS_STRIDE];
            // B layout: lanes 0-15 K={0..7,16..23} of tile, lanes 16-31 K={8..15,24..31}
            v8bf b0lo = *(const v8bf*)(src + 8 * lhalf);
            v8bf b0hi = *(const v8bf*)(src + 16 + 8 * lhalf);
            v8bf b1lo = *(const v8bf*)(src + 32 + 8 * lhalf);
            v8bf b1hi = *(const v8bf*)(src + 48 + 8 * lhalf);
            v16bf b0 = __builtin_shufflevector(b0lo, b0hi, 0,1,2,3,4,5,6,7,8,9,10,11,12,13,14,15);
            v16bf b1 = __builtin_shufflevector(b1lo, b1hi, 0,1,2,3,4,5,6,7,8,9,10,11,12,13,14,15);

            v8f acc = {};
            acc = __builtin_amdgcn_wmma_f32_16x16x32_bf16(false, a0, false, b0,
                                                          (short)0, acc, false, false);
            acc = __builtin_amdgcn_wmma_f32_16x16x32_bf16(false, a1, false, b1,
                                                          (short)0, acc, false, false);

            const float cn   = cnormS[lr];
            const int   gcol = p * PHASE_ROWS + lr;
#pragma unroll
            for (int r = 0; r < 8; ++r) {         // rows m = 8*lhalf + r
                float s = cn - 2.0f * acc[r];     // ||x||^2 omitted: row-constant
                if (s < bestV[r]) { bestV[r] = s; bestI[r] = gcol; }
            }
        }
    }

    // ---- argmin reduction across the 16 lanes holding each row ----
#pragma unroll
    for (int off = 1; off < 16; off <<= 1) {
#pragma unroll
        for (int r = 0; r < 8; ++r) {
            float vv = __shfl_xor(bestV[r], off, 32);
            int   ii = __shfl_xor(bestI[r], off, 32);
            if (vv < bestV[r] || (vv == bestV[r] && ii < bestI[r])) {
                bestV[r] = vv; bestI[r] = ii;
            }
        }
    }
    if (l15 == 0) {
#pragma unroll
        for (int r = 0; r < 8; ++r)
            idxS[wave * 16 + lhalf * 8 + r] = bestI[r];
    }
    __syncthreads();

    // ---- one-hot output: coalesced non-temporal float4 streaming ----
    // 512 MB of write-once data: NT stores rinse through L2 so the codebook
    // stays resident for the gather phase below.
    {
        v4f* dbase = (v4f*)(disc + (size_t)rowBase * KCB);
#pragma unroll 4
        for (int i = 0; i < (ROWS_PER_WG * KCB / 4) / 256; ++i) {  // 128 iters
            int flat = t + 256 * i;
            int row  = flat >> 8;       // 256 float4 per row
            int col4 = flat & 255;
            int idx  = idxS[row];
            v4f v = {0.f, 0.f, 0.f, 0.f};
            if ((idx >> 2) == col4) v[idx & 3] = 1.0f;
            __builtin_nontemporal_store(v, &dbase[flat]);
        }
    }
    // ---- quantized output: gather exact fp32 codebook rows (L2-resident) ----
    {
        v4f* qbase = (v4f*)(quant + (size_t)rowBase * DIM);
        const v4f* cb4 = (const v4f*)cb;
#pragma unroll
        for (int i = 0; i < (ROWS_PER_WG * DIM / 4) / 256; ++i) {  // 8 iters
            int flat = t + 256 * i;
            int row  = flat >> 4;       // 16 float4 per row
            int c4   = flat & 15;
            v4f v = cb4[(size_t)idxS[row] * 16 + c4];
            __builtin_nontemporal_store(v, &qbase[flat]);
        }
    }
}

extern "C" void kernel_launch(void* const* d_in, const int* in_sizes, int n_in,
                              void* d_out, int out_size, void* d_ws, size_t ws_size,
                              hipStream_t stream) {
    const float* x  = (const float*)d_in[0];
    const float* cb = (const float*)d_in[1];
    const int n = in_sizes[0] / DIM;           // 131072 rows
    float* disc  = (float*)d_out;              // [n, 1024]
    float* quant = disc + (size_t)n * KCB;     // [n, 64]
    vq_wmma_kernel<<<n / ROWS_PER_WG, 256, 0, stream>>>(x, cb, disc, quant);
}
